// SpatialRelationshipAttention_29308856828241
// MI455X (gfx1250) — matrix operand order repaired and checked by reading, not compiled, not run on hardware
//
#include <hip/hip_runtime.h>

typedef __attribute__((ext_vector_type(16))) _Float16 v16h;
typedef __attribute__((ext_vector_type(8)))  float    v8f;

union AF { v16h v; _Float16 e[16]; };
union CF { v8f  v; float    e[8];  };

#define DEVI __device__ __forceinline__

constexpr int Bb = 8, S = 1024, D = 256, NH = 8, DK = 32;
constexpr float SCALE = 0.17677669529663687f;  // 1/sqrt(32)
constexpr float PI_F  = 3.14159265358979323846f;

DEVI float half_max(float v) {
#pragma unroll
  for (int m = 1; m < 16; m <<= 1) v = fmaxf(v, __shfl_xor(v, m, 32));
  return v;
}
DEVI float half_sum(float v) {
#pragma unroll
  for (int m = 1; m < 16; m <<= 1) v += __shfl_xor(v, m, 32);
  return v;
}
DEVI v8f wmma16(v16h a, v16h b, v8f c) {
  // V_WMMA_F32_16X16X32_F16 : D = A(16x32 f16) x B(32x16 f16) + C(f32)
  return __builtin_amdgcn_wmma_f32_16x16x32_f16(false, a, false, b, (short)0, c,
                                                false, false);
}

// --- CDNA5 async copy: global -> LDS, tracked by ASYNCcnt -------------------
DEVI void async_ld_b128(unsigned lds_byte_off, const void* gsrc) {
  // GLOBAL_LOAD_ASYNC_TO_LDS_B128 (GV mode): per-lane 16B, 512B per wave.
  asm volatile("global_load_async_to_lds_b128 %0, %1, off"
               :: "v"(lds_byte_off), "v"(gsrc)
               : "memory");
}
DEVI void wait_async_le2() {
#if __has_builtin(__builtin_amdgcn_s_wait_asynccnt)
  __builtin_amdgcn_s_wait_asynccnt(2);
#else
  asm volatile("s_wait_asynccnt 0x2" ::: "memory");
#endif
}
DEVI void wait_async_le0() {
#if __has_builtin(__builtin_amdgcn_s_wait_asynccnt)
  __builtin_amdgcn_s_wait_asynccnt(0);
#else
  asm volatile("s_wait_asynccnt 0x0" ::: "memory");
#endif
}

// ---------------------------------------------------------------------------
// Kernel 0: packed spatial index table pidx[sq*S+sk] = dist*16 + dir  (u16)
// ---------------------------------------------------------------------------
__global__ void k_pidx(unsigned short* __restrict__ pidx) {
  int idx = blockIdx.x * 256 + threadIdx.x;            // [0, S*S)
  int sq = idx >> 10, sk = idx & 1023;
  int hq = sq >> 5, wq = sq & 31, hk = sk >> 5, wk = sk & 31;
  float dh = (float)(hk - hq), dw = (float)(wk - wq);
  int dist = (int)floorf(sqrtf(dh * dh + dw * dw));
  dist = dist > 59 ? 59 : dist;
  int dirn = 0;
  if (!((hk == hq) && (wk == wq))) {
    float ang = atan2f(dh, dw) + PI_F;                 // [0, 2pi]
    dirn = ((int)floorf(ang * (16.0f / (2.0f * PI_F)))) & 15;
  }
  pidx[idx] = (unsigned short)(dist * 16 + dirn);
}

// ---------------------------------------------------------------------------
// Kernel 1: fused QKV projection, f16 WMMA GEMM  out = x @ W^T + b
//   q,k -> f16 [b][h][s][32] row-major ; v -> f16 transposed [b][h][32][s]
// One wave per 16x16 output tile; 8 WMMAs over K=256.
// ---------------------------------------------------------------------------
__global__ void k_qkv(const float* __restrict__ x,
                      const float* __restrict__ wq, const float* __restrict__ bq,
                      const float* __restrict__ wk, const float* __restrict__ bk,
                      const float* __restrict__ wv, const float* __restrict__ bv,
                      _Float16* __restrict__ qf, _Float16* __restrict__ kf,
                      _Float16* __restrict__ vt) {
  int gid  = blockIdx.x * 8 + (threadIdx.x >> 5);  // 3 * 512 * 16 tiles
  int lane = threadIdx.x & 31;
  int mat  = gid >> 13;                            // 0=q 1=k 2=v (8192 tiles ea.)
  int rem  = gid & 8191;
  int rt = rem >> 4, ct = rem & 15;
  const float* Wm = (mat == 0) ? wq : (mat == 1) ? wk : wv;
  const float* bm = (mat == 0) ? bq : (mat == 1) ? bk : bv;
  int nl = lane & 15;
  int ab = (lane >> 4) << 3;                       // A half-wave K offset (+8)
  int bbo = (lane >> 4) << 4;                      // unused; B loads also use ab (+8)
  int arow = rt * 16 + nl;
  int brow = ct * 16 + nl;
  CF acc;
#pragma unroll
  for (int r = 0; r < 8; ++r) acc.e[r] = 0.f;
  for (int kk = 0; kk < D; kk += 32) {
    AF a, bf;
    const float* xa = x + (long)arow * D + kk;
    __builtin_prefetch(xa + 32, 0, 0);
#pragma unroll
    for (int h = 0; h < 8; ++h) a.e[h] = (_Float16)xa[ab + h];
#pragma unroll
    for (int h = 0; h < 8; ++h) a.e[8 + h] = (_Float16)xa[16 + ab + h];
    const float* wb = Wm + (long)brow * D + kk;
#pragma unroll
    for (int h = 0; h < 8; ++h) bf.e[h] = (_Float16)wb[ab + h];
#pragma unroll
    for (int h = 0; h < 8; ++h) bf.e[8 + h] = (_Float16)wb[16 + ab + h];
    acc.v = wmma16(a.v, bf.v, acc.v);
  }
  int n = ct * 16 + nl;
  float bias = bm[n];
  int hd = n >> 5, dd = n & 31;
#pragma unroll
  for (int r = 0; r < 8; ++r) {
    int grow = rt * 16 + r + ((lane >> 4) << 3);   // C layout: m = r + 8*(lane>>4)
    int b = grow >> 10, s = grow & 1023;
    _Float16 val = (_Float16)(acc.e[r] + bias);
    long bh = (long)(b * NH + hd);
    if (mat == 0)      qf[(bh * S + s) * DK + dd] = val;
    else if (mat == 1) kf[(bh * S + s) * DK + dd] = val;
    else               vt[(bh * DK + dd) * S + s] = val;
  }
}

// ---------------------------------------------------------------------------
// Kernel 2: fused attention per (b, 16-query tile). 8 waves = 8 heads.
//   Pass 1: row max/sum stats (biased + unbiased); K tiles double-buffered in
//           LDS via GLOBAL_LOAD_ASYNC_TO_LDS_B128 + s_wait_asynccnt.
//   Pass 2: recompute scores; attn_w (head mean, unbiased) via ds_add_f32
//           reduction; ctx accumulated with WMMA (P staged through LDS).
// ---------------------------------------------------------------------------
__global__ void k_attn(const _Float16* __restrict__ qf,
                       const _Float16* __restrict__ kf,
                       const _Float16* __restrict__ vt,
                       const unsigned short* __restrict__ pidx,
                       const float* __restrict__ dist_bias,
                       const float* __restrict__ dir_bias,
                       float* __restrict__ attn_out,
                       _Float16* __restrict__ ctxf) {
  __shared__ float     cb[960 * NH];          // combined (dist,dir) bias LUT
  __shared__ _Float16  pstage[NH * 16 * 32];  // per-wave P tile (C->A relayout)
  __shared__ float     attacc[16 * 32];       // head-mean accumulation
  __shared__ _Float16  kstage[NH * 2 * 16 * DK];  // async K tiles, dbl-buffered

  int tid = threadIdx.x, lane = tid & 31, head = tid >> 5;
  int b = blockIdx.x >> 6, qt = blockIdx.x & 63;

  for (int i = tid; i < 960 * NH; i += 256) {
    int pr = i / NH, h = i % NH;
    int di = pr >> 4, dr = pr & 15;
    cb[i] = dist_bias[di * NH + h] + dir_bias[dr * NH + h];
  }
  __syncthreads();

  long bh = (long)(b * NH + head);
  const _Float16* qbase = qf + bh * S * DK;
  const _Float16* kbase = kf + bh * S * DK;
  const _Float16* vbase = vt + bh * DK * S;
  const unsigned short* pq = pidx + (long)(qt * 16) * S;

  int nl = lane & 15;
  int mh = (lane >> 4) << 3;   // C-layout row offset of this half-wave
  int ab = (lane >> 4) << 3;   // A-fragment K offset
  int bbo = (lane >> 4) << 4;  // unused; B fragments also use ab

  AF aq;
  {
    const _Float16* qa = qbase + (qt * 16 + nl) * DK;
#pragma unroll
    for (int h = 0; h < 8; ++h) aq.e[h] = qa[ab + h];
#pragma unroll
    for (int h = 0; h < 8; ++h) aq.e[8 + h] = qa[16 + ab + h];
  }

  float row_mb[8], row_lb[8], row_mu[8], row_lu[8];
#pragma unroll
  for (int r = 0; r < 8; ++r) {
    row_mb[r] = -3.4e38f; row_lb[r] = 0.f;
    row_mu[r] = -3.4e38f; row_lu[r] = 0.f;
  }

  // ---- pass 1: statistics over all 64 key tiles --------------------------
  // Each 16-key tile of K is a contiguous 1KB block; stage it into LDS with
  // two async b128 copies per wave (lane L covers bytes 16L and 512+16L).
  const _Float16* kst = kstage + head * (2 * 16 * DK);
  unsigned kst_base = (unsigned)(unsigned long long)(const void*)kst; // LDS off
  {
    const _Float16* src = kbase + (long)0 * 16 * DK + lane * 8;
    unsigned dst = kst_base + (unsigned)(lane * 16);
    async_ld_b128(dst, src);
    async_ld_b128(dst + 512u, src + 256);
  }
  for (int kt = 0; kt < 64; ++kt) {
    if (kt < 63) {
      int nb = (kt + 1) & 1;
      const _Float16* src = kbase + (long)(kt + 1) * 16 * DK + lane * 8;
      unsigned dst = kst_base + (unsigned)(nb * 1024 + lane * 16);
      async_ld_b128(dst, src);
      async_ld_b128(dst + 512u, src + 256);
      wait_async_le2();   // the 2 in-flight are next tile's; current is done
    } else {
      wait_async_le0();
    }
    AF bk_;
    const _Float16* ka = kst + (kt & 1) * (16 * DK) + nl * DK;
#pragma unroll
    for (int h = 0; h < 8; ++h) bk_.e[h] = ka[ab + h];    // ds_load from LDS
#pragma unroll
    for (int h = 0; h < 8; ++h) bk_.e[8 + h] = ka[16 + ab + h];
    CF c;
#pragma unroll
    for (int r = 0; r < 8; ++r) c.e[r] = 0.f;
    c.v = wmma16(aq.v, bk_.v, c.v);
    int sk = kt * 16 + nl;
#pragma unroll
    for (int r = 0; r < 8; ++r) {
      float su = c.e[r] * SCALE;
      float sb = su + cb[(int)pq[(r + mh) * S + sk] * NH + head];
      float mb = fmaxf(row_mb[r], half_max(sb));
      float mu = fmaxf(row_mu[r], half_max(su));
      row_lb[r] = row_lb[r] * __expf(row_mb[r] - mb) + half_sum(__expf(sb - mb));
      row_lu[r] = row_lu[r] * __expf(row_mu[r] - mu) + half_sum(__expf(su - mu));
      row_mb[r] = mb; row_mu[r] = mu;
    }
  }
  float row_rlb[8], row_rlu[8];
#pragma unroll
  for (int r = 0; r < 8; ++r) {
    row_rlb[r] = 1.f / row_lb[r];
    row_rlu[r] = 0.125f / row_lu[r];   // /8 heads folded in
  }

  // ---- pass 2: probs, attn_w output, ctx accumulation --------------------
  CF ctx0, ctx1;
#pragma unroll
  for (int r = 0; r < 8; ++r) { ctx0.e[r] = 0.f; ctx1.e[r] = 0.f; }
  _Float16* ps = pstage + head * 16 * 32;

  for (int kb = 0; kb < 32; ++kb) {     // key blocks of 32
    __syncthreads();
    attacc[tid] = 0.f; attacc[tid + 256] = 0.f;
    __syncthreads();
#pragma unroll
    for (int ct = 0; ct < 2; ++ct) {
      int kt = kb * 2 + ct;
      AF bk_;
      const _Float16* ka = kbase + (kt * 16 + nl) * DK;
#pragma unroll
      for (int h = 0; h < 8; ++h) bk_.e[h] = ka[ab + h];
#pragma unroll
      for (int h = 0; h < 8; ++h) bk_.e[8 + h] = ka[16 + ab + h];
      CF c;
#pragma unroll
      for (int r = 0; r < 8; ++r) c.e[r] = 0.f;
      c.v = wmma16(aq.v, bk_.v, c.v);
      int sk = kt * 16 + nl;
#pragma unroll
      for (int r = 0; r < 8; ++r) {
        float su = c.e[r] * SCALE;
        float bias = cb[(int)pq[(r + mh) * S + sk] * NH + head];
        float pb = __expf(su + bias - row_mb[r]) * row_rlb[r];
        float pu = __expf(su - row_mu[r]) * row_rlu[r];
        int m = r + mh;
        ps[m * 32 + ct * 16 + nl] = (_Float16)pb;
        atomicAdd(&attacc[m * 32 + ct * 16 + nl], pu);  // ds_add_f32
      }
    }
    // relayout P (C-layout in LDS) into A-fragment; same-wave LDS is in-order
    AF ap;
    {
      const _Float16* pa = ps + nl * 32;
#pragma unroll
      for (int h = 0; h < 8; ++h) ap.e[h] = pa[ab + h];
#pragma unroll
      for (int h = 0; h < 8; ++h) ap.e[8 + h] = pa[16 + ab + h];
    }
#pragma unroll
    for (int dt = 0; dt < 2; ++dt) {
      AF bv;
      const _Float16* va = vbase + (dt * 16 + nl) * S + kb * 32;
#pragma unroll
      for (int h = 0; h < 8; ++h) bv.e[h] = va[ab + h];
#pragma unroll
      for (int h = 0; h < 8; ++h) bv.e[8 + h] = va[16 + ab + h];
      if (dt == 0) ctx0.v = wmma16(ap.v, bv.v, ctx0.v);
      else         ctx1.v = wmma16(ap.v, bv.v, ctx1.v);
    }
    __syncthreads();
    {
      long rowbase = (long)b * S + qt * 16;
      int i0 = tid, i1 = tid + 256;
      attn_out[(rowbase + (i0 >> 5)) * S + kb * 32 + (i0 & 31)] = attacc[i0];
      attn_out[(rowbase + (i1 >> 5)) * S + kb * 32 + (i1 & 31)] = attacc[i1];
    }
  }
  // store ctx (already normalized) as f16 [b*S+s][256] head-major
#pragma unroll
  for (int r = 0; r < 8; ++r) {
    int m = r + mh;
    long srow = (long)b * S + qt * 16 + m;
    ctxf[srow * D + head * 32 + nl]      = (_Float16)ctx0.e[r];
    ctxf[srow * D + head * 32 + 16 + nl] = (_Float16)ctx1.e[r];
  }
}

// ---------------------------------------------------------------------------
// Kernel 3: out = ctx @ wo^T + bo ; residual ; LayerNorm -> y
// 16 waves cover 16 rows x 256 cols; LDS-atomic row reductions for LN.
// ---------------------------------------------------------------------------
__global__ void k_out(const _Float16* __restrict__ ctxf,
                      const float* __restrict__ x,
                      const float* __restrict__ wo, const float* __restrict__ bo,
                      const float* __restrict__ ln_g, const float* __restrict__ ln_b,
                      float* __restrict__ y) {
  __shared__ float s_sum[16], s_sq[16];
  int tid = threadIdx.x, lane = tid & 31, wvid = tid >> 5;  // wave = col tile
  int rt = blockIdx.x;
  int nl = lane & 15;
  int mh = (lane >> 4) << 3;
  int ab = (lane >> 4) << 3;
  int bbo = (lane >> 4) << 4;
  if (tid < 16) { s_sum[tid] = 0.f; s_sq[tid] = 0.f; }
  CF acc;
#pragma unroll
  for (int r = 0; r < 8; ++r) acc.e[r] = 0.f;
  int arow = rt * 16 + nl;
  for (int kk = 0; kk < D; kk += 32) {
    AF a, bf;
    const _Float16* ca = ctxf + (long)arow * D + kk;
#pragma unroll
    for (int h = 0; h < 8; ++h) a.e[h] = ca[ab + h];
#pragma unroll
    for (int h = 0; h < 8; ++h) a.e[8 + h] = ca[16 + ab + h];
    const float* wb = wo + (long)(wvid * 16 + nl) * D + kk;
#pragma unroll
    for (int h = 0; h < 8; ++h) bf.e[h] = (_Float16)wb[ab + h];
#pragma unroll
    for (int h = 0; h < 8; ++h) bf.e[8 + h] = (_Float16)wb[16 + ab + h];
    acc.v = wmma16(a.v, bf.v, acc.v);
  }
  __syncthreads();
  int col = wvid * 16 + nl;
  float yp[8];
#pragma unroll
  for (int r = 0; r < 8; ++r) {
    int grow = rt * 16 + r + mh;
    yp[r] = acc.e[r] + bo[col] + x[(long)grow * D + col];
    float rs = half_sum(yp[r]);
    float rq = half_sum(yp[r] * yp[r]);
    if (nl == 0) {
      atomicAdd(&s_sum[r + mh], rs);
      atomicAdd(&s_sq[r + mh], rq);
    }
  }
  __syncthreads();
  float gg = ln_g[col], bb2 = ln_b[col];
#pragma unroll
  for (int r = 0; r < 8; ++r) {
    int m = r + mh;
    float mean = s_sum[m] * (1.f / 256.f);
    float var  = s_sq[m] * (1.f / 256.f) - mean * mean;
    float inv  = rsqrtf(var + 1e-5f);
    int grow = rt * 16 + m;
    y[(long)grow * D + col] = (yp[r] - mean) * inv * gg + bb2;
  }
}

// ---------------------------------------------------------------------------
extern "C" void kernel_launch(void* const* d_in, const int* in_sizes, int n_in,
                              void* d_out, int out_size, void* d_ws, size_t ws_size,
                              hipStream_t stream) {
  const float* x    = (const float*)d_in[0];
  const float* wq   = (const float*)d_in[1];
  const float* bq   = (const float*)d_in[2];
  const float* wk   = (const float*)d_in[3];
  const float* bk   = (const float*)d_in[4];
  const float* wvw  = (const float*)d_in[5];
  const float* bv   = (const float*)d_in[6];
  const float* wo   = (const float*)d_in[7];
  const float* bo   = (const float*)d_in[8];
  const float* lng  = (const float*)d_in[9];
  const float* lnb  = (const float*)d_in[10];
  const float* dist = (const float*)d_in[11];
  const float* dirb = (const float*)d_in[12];

  char* ws = (char*)d_ws;
  unsigned short* pidx = (unsigned short*)ws;              //  2 MB
  _Float16* qf  = (_Float16*)(ws + (2l  << 20));           //  4 MB
  _Float16* kf  = (_Float16*)(ws + (6l  << 20));           //  4 MB
  _Float16* vt  = (_Float16*)(ws + (10l << 20));           //  4 MB
  _Float16* ctx = (_Float16*)(ws + (14l << 20));           //  4 MB

  float* y    = (float*)d_out;                             // [B,H,W,D]
  float* attn = y + (size_t)Bb * S * D;                    // [B,S,S]

  k_pidx<<<(S * S) / 256, 256, 0, stream>>>(pidx);
  k_qkv <<<3 * 512 * 16 / 8, 256, 0, stream>>>(x, wq, bq, wk, bk, wvw, bv,
                                               qf, kf, vt);
  k_attn<<<Bb * (S / 16), 256, 0, stream>>>(qf, kf, vt, pidx, dist, dirb,
                                            attn, ctx);
  k_out <<<Bb * (S / 16), 512, 0, stream>>>(ctx, x, wo, bo, lng, lnb, y);
}
